// LucaGPLMMultiheadAttention_11751030522568
// MI455X (gfx1250) — matrix operand order repaired and checked by reading, not compiled, not run on hardware
//
#include <hip/hip_runtime.h>

// ---------------------------------------------------------------------------
// Types
// ---------------------------------------------------------------------------
typedef __bf16 bf16_t;
typedef __attribute__((ext_vector_type(16))) __bf16 v16bf;
typedef __attribute__((ext_vector_type(8)))  __bf16 v8bf;
typedef __attribute__((ext_vector_type(8)))  float  v8f;

#define EMBED_DIM 1024
#define NUM_HEADS 16
#define HEAD_DIM  64
#define SEQ_LEN   2048
#define BATCH     2
#define MTOT      (SEQ_LEN * BATCH)   // 4096 rows

// ---------------------------------------------------------------------------
// bf16 helpers (round-to-nearest-even)
// ---------------------------------------------------------------------------
__device__ __forceinline__ bf16_t f2bf(float f) {
  unsigned u = __builtin_bit_cast(unsigned, f);
  u += 0x7FFFu + ((u >> 16) & 1u);
  unsigned short h = (unsigned short)(u >> 16);
  return __builtin_bit_cast(bf16_t, h);
}
__device__ __forceinline__ float bf2f(bf16_t b) {
  unsigned short h = __builtin_bit_cast(unsigned short, b);
  unsigned u = ((unsigned)h) << 16;
  return __builtin_bit_cast(float, u);
}

__device__ __forceinline__ v16bf cat16(v8bf lo, v8bf hi) {
  v16bf r;
#pragma unroll
  for (int i = 0; i < 8; ++i) { r[i] = lo[i]; r[i + 8] = hi[i]; }
  return r;
}

// ---------------------------------------------------------------------------
// WMMA fragment loaders (wave32).
// A-matrix 16x32 bf16 (ISA 7.12.2): lanes 0-15: M=lane, VGPR0-3 K=0..7,
// VGPR4-7 K=16..23; lanes 16-31: M=lane-16, K=8..15 / 24..31.
// Source must be row-major with contiguous K (leading dim = ld elements).
// ---------------------------------------------------------------------------
__device__ __forceinline__ v16bf load_fragA(const bf16_t* base, int ld, int kk) {
  int lane = threadIdx.x & 31;
  int m  = lane & 15;
  int kb = kk + ((lane >> 4) << 3);          // +8 for upper lane half
  const bf16_t* p = base + (size_t)m * ld + kb;
  v8bf lo = *reinterpret_cast<const v8bf*>(p);        // K = kb..kb+7
  v8bf hi = *reinterpret_cast<const v8bf*>(p + 16);   // K = kb+16..kb+23
  return cat16(lo, hi);
}

// B-matrix 32x16 bf16 (same K order as A): lanes 0-15 hold
// K=0..7 / 16..23, lanes 16-31 hold K=8..15 / 24..31; column N = lane&15.
// base points at row-major [N][K] storage (i.e. B[k][n] = base[n*ld + k]).
__device__ __forceinline__ v16bf load_fragB(const bf16_t* base, int ld, int kk) {
  int lane = threadIdx.x & 31;
  int n  = lane & 15;
  int kb = kk + ((lane >> 4) << 3);          // +8 for upper lane half
  const bf16_t* p = base + (size_t)n * ld + kb;
  v8bf lo = *reinterpret_cast<const v8bf*>(p);        // K = kb..kb+7
  v8bf hi = *reinterpret_cast<const v8bf*>(p + 16);   // K = kb+16..kb+23
  return cat16(lo, hi);
}

__device__ __forceinline__ v8f wmma_bf16(v16bf a, v16bf b, v8f c) {
  return __builtin_amdgcn_wmma_f32_16x16x32_bf16(
      /*neg_a=*/false, a, /*neg_b=*/false, b,
      /*c_mod=*/(short)0, c, /*reuse_a=*/false, /*reuse_b=*/false);
}

// ---------------------------------------------------------------------------
// Stage 0: fp32 -> bf16 conversion (grid-stride)
// ---------------------------------------------------------------------------
__global__ void k_f32_to_bf16(const float* __restrict__ in,
                              bf16_t* __restrict__ out, int n) {
  for (int i = blockIdx.x * blockDim.x + threadIdx.x; i < n;
       i += gridDim.x * blockDim.x)
    out[i] = f2bf(in[i]);
}

// ---------------------------------------------------------------------------
// Stage 1/4: GEMM  C[M][N] = alpha * (A[M][K] @ W[N][K]^T + bias[N])
// Block: 256 threads = 8 waves, tile 128x128. Wave tile 32x64.
// OUT_F32: write fp32 (final projection) else bf16.
// ---------------------------------------------------------------------------
template <bool OUT_F32>
__global__ __launch_bounds__(256)
void k_gemm_bf16(const bf16_t* __restrict__ A, const bf16_t* __restrict__ W,
                 const float* __restrict__ bias, void* __restrict__ Cout,
                 int M, int N, int K, float alpha) {
  const int wave = threadIdx.x >> 5;
  const int lane = threadIdx.x & 31;
  const int m0 = blockIdx.x * 128 + (wave >> 1) * 32;
  const int n0 = blockIdx.y * 128 + (wave & 1) * 64;

  v8f acc[2][4];
#pragma unroll
  for (int i = 0; i < 2; ++i)
#pragma unroll
    for (int j = 0; j < 4; ++j) acc[i][j] = (v8f)0.0f;

  for (int kk = 0; kk < K; kk += 32) {
    v16bf a0 = load_fragA(A + (size_t)m0 * K, K, kk);
    v16bf a1 = load_fragA(A + (size_t)(m0 + 16) * K, K, kk);
#pragma unroll
    for (int j = 0; j < 4; ++j) {
      v16bf b = load_fragB(W + (size_t)(n0 + j * 16) * K, K, kk);
      acc[0][j] = wmma_bf16(a0, b, acc[0][j]);
      acc[1][j] = wmma_bf16(a1, b, acc[1][j]);
    }
  }

  // Epilogue. C layout: VGPR r -> row (r + 8*(lane>>4)), col = lane&15.
  const int rowOff = (lane >> 4) * 8;
  const int colLane = lane & 15;
#pragma unroll
  for (int mi = 0; mi < 2; ++mi) {
#pragma unroll
    for (int j = 0; j < 4; ++j) {
      const int col = n0 + j * 16 + colLane;
      const float bv = bias[col];
#pragma unroll
      for (int r = 0; r < 8; ++r) {
        const int row = m0 + mi * 16 + rowOff + r;
        const float v = alpha * (acc[mi][j][r] + bv);
        if (OUT_F32)
          ((float*)Cout)[(size_t)row * N + col] = v;
        else
          ((bf16_t*)Cout)[(size_t)row * N + col] = f2bf(v);
      }
    }
  }
}

// ---------------------------------------------------------------------------
// Stage 2: RoPE + scatter into per-head layouts.
//   qh/kh : [B*H][S][hd]   (rope applied; q already has hd^-0.5 scaling)
//   vT    : [B*H][hd][S]   (V transposed so PV B-fragments are contiguous-K)
// ---------------------------------------------------------------------------
__global__ __launch_bounds__(256)
void k_rope_scatter(const bf16_t* __restrict__ qt, const bf16_t* __restrict__ kt,
                    const bf16_t* __restrict__ vt,
                    bf16_t* __restrict__ qh, bf16_t* __restrict__ kh,
                    bf16_t* __restrict__ vT) {
  const int hd = HEAD_DIM, H = NUM_HEADS, B = BATCH, S = SEQ_LEN;
  int idx = blockIdx.x * blockDim.x + threadIdx.x;
  const int total = S * B * H * hd;
  if (idx >= total) return;

  const int d = idx & (hd - 1);
  int t = idx >> 6;              // hd == 64
  const int h = t % H; t /= H;
  const int b = t % B;
  const int s = t / B;

  const size_t row = ((size_t)s * B + b) * EMBED_DIM + (size_t)h * hd;
  const float q = bf2f(qt[row + d]);
  const float k = bf2f(kt[row + d]);
  const float v = bf2f(vt[row + d]);

  const int j = d & 31;                               // freq index
  const float ang = (float)s * __powf(10000.0f, -(float)(2 * j) / (float)hd);
  float sn, cs;
  __sincosf(ang, &sn, &cs);

  const int dp = (d < 32) ? d + 32 : d - 32;          // rotate_half partner
  const float qp = bf2f(qt[row + dp]);
  const float kp = bf2f(kt[row + dp]);
  const float rq = (d < 32) ? -qp : qp;
  const float rk = (d < 32) ? -kp : kp;

  const float qo = q * cs + rq * sn;
  const float ko = k * cs + rk * sn;

  const int bh = b * H + h;
  const size_t dst = ((size_t)bh * S + s) * hd + d;
  qh[dst] = f2bf(qo);
  kh[dst] = f2bf(ko);
  vT[((size_t)bh * hd + d) * S + s] = f2bf(v);
}

// ---------------------------------------------------------------------------
// Stage 3: flash attention. Grid = (B*H, S/128), 256 threads = 8 waves.
// One wave owns 16 query rows; streams keys in blocks of 64 with online
// softmax. P goes through a per-wave LDS tile to convert C-layout -> A-layout.
// ---------------------------------------------------------------------------
__global__ __launch_bounds__(256)
void k_flash_attn(const bf16_t* __restrict__ qh, const bf16_t* __restrict__ kh,
                  const bf16_t* __restrict__ vT, bf16_t* __restrict__ ctx) {
  const int hd = HEAD_DIM, H = NUM_HEADS, B = BATCH, S = SEQ_LEN;
  const int bh = blockIdx.x;
  const int wave = threadIdx.x >> 5;
  const int lane = threadIdx.x & 31;
  const int s0 = blockIdx.y * 128 + wave * 16;

  const bf16_t* Q  = qh + (size_t)bh * S * hd + (size_t)s0 * hd;
  const bf16_t* Kh = kh + (size_t)bh * S * hd;
  const bf16_t* Vt = vT + (size_t)bh * hd * S;

  __shared__ __align__(16) bf16_t pshm[8][16][64];   // per-wave P tile, 16 KB

  v8f acc[4];
#pragma unroll
  for (int j = 0; j < 4; ++j) acc[j] = (v8f)0.0f;
  float mrow[8], lrow[8];
#pragma unroll
  for (int r = 0; r < 8; ++r) { mrow[r] = -1e30f; lrow[r] = 0.0f; }

  for (int kb = 0; kb < S; kb += 64) {
    // ---- S-tile = Q(16x64) @ K^T(64x64) -----------------------------------
    v8f sfr[4];
#pragma unroll
    for (int j = 0; j < 4; ++j) sfr[j] = (v8f)0.0f;
#pragma unroll
    for (int kk = 0; kk < hd; kk += 32) {
      v16bf a = load_fragA(Q, hd, kk);
#pragma unroll
      for (int j = 0; j < 4; ++j) {
        v16bf b = load_fragB(Kh + (size_t)(kb + j * 16) * hd, hd, kk);
        sfr[j] = wmma_bf16(a, b, sfr[j]);
      }
    }

    // ---- online softmax (rows live in 16-lane groups of each VGPR) -------
#pragma unroll
    for (int r = 0; r < 8; ++r) {
      float mx = fmaxf(fmaxf(sfr[0][r], sfr[1][r]), fmaxf(sfr[2][r], sfr[3][r]));
#pragma unroll
      for (int off = 1; off < 16; off <<= 1) mx = fmaxf(mx, __shfl_xor(mx, off, 32));
      const float mnew = fmaxf(mrow[r], mx);
      const float scale = __expf(mrow[r] - mnew);
      float sum = 0.0f;
#pragma unroll
      for (int j = 0; j < 4; ++j) {
        float p = __expf(sfr[j][r] - mnew);
        sfr[j][r] = p;
        sum += p;
      }
#pragma unroll
      for (int off = 1; off < 16; off <<= 1) sum += __shfl_xor(sum, off, 32);
      lrow[r] = lrow[r] * scale + sum;
      mrow[r] = mnew;
#pragma unroll
      for (int j = 0; j < 4; ++j) acc[j][r] *= scale;
    }

    // ---- P: C-layout regs -> LDS (row-major 16x64 bf16, per-wave) --------
    {
      const int rowOff = (lane >> 4) * 8;
      const int col = lane & 15;
#pragma unroll
      for (int r = 0; r < 8; ++r)
#pragma unroll
        for (int j = 0; j < 4; ++j)
          pshm[wave][rowOff + r][j * 16 + col] = f2bf(sfr[j][r]);
      // same-wave LDS ops are in-order; compiler inserts s_wait_dscnt before
      // the dependent ds_loads below.
    }

    // ---- acc += P(16x64) @ V(64x64)  (V read transposed: Vt[hd][S]) ------
#pragma unroll
    for (int kk = 0; kk < 64; kk += 32) {
      v16bf a = load_fragA(&pshm[wave][0][0], 64, kk);
#pragma unroll
      for (int j = 0; j < 4; ++j) {
        v16bf b = load_fragB(Vt + (size_t)(j * 16) * S + kb, S, kk);
        acc[j] = wmma_bf16(a, b, acc[j]);
      }
    }
  }

  // ---- normalize & scatter to context [S*B][E] (row = s*B+b, col = h*hd+d)
  const int b = bh / H, h = bh % H;
  const int rowOff = (lane >> 4) * 8;
  const int cl = lane & 15;
#pragma unroll
  for (int r = 0; r < 8; ++r) {
    const int s = s0 + rowOff + r;
    const float inv = 1.0f / lrow[r];
    const size_t base = ((size_t)s * B + b) * EMBED_DIM + (size_t)h * hd;
#pragma unroll
    for (int j = 0; j < 4; ++j)
      ctx[base + j * 16 + cl] = f2bf(acc[j][r] * inv);
  }
}

// ---------------------------------------------------------------------------
// Host-side launcher
// ---------------------------------------------------------------------------
extern "C" void kernel_launch(void* const* d_in, const int* in_sizes, int n_in,
                              void* d_out, int out_size, void* d_ws, size_t ws_size,
                              hipStream_t stream) {
  (void)in_sizes; (void)n_in; (void)out_size; (void)ws_size;

  const int E = EMBED_DIM, M = MTOT, S = SEQ_LEN, B = BATCH, H = NUM_HEADS;
  const float scaling = 0.125f;   // HEAD_DIM^-0.5 = 1/8

  const float* query = (const float*)d_in[0];
  const float* Wq = (const float*)d_in[1]; const float* bq = (const float*)d_in[2];
  const float* Wk = (const float*)d_in[3]; const float* bk = (const float*)d_in[4];
  const float* Wv = (const float*)d_in[5]; const float* bv = (const float*)d_in[6];
  const float* Wo = (const float*)d_in[7]; const float* bo = (const float*)d_in[8];

  // ---- workspace carve-out (bf16 buffers), ~72 MB total -------------------
  char* ws = (char*)d_ws;
  size_t off = 0;
  auto carve = [&](size_t elems) { bf16_t* p = (bf16_t*)(ws + off); off += elems * sizeof(bf16_t); return p; };
  bf16_t* xbf  = carve((size_t)M * E);        // query in bf16
  bf16_t* wqb  = carve((size_t)E * E);
  bf16_t* wkb  = carve((size_t)E * E);
  bf16_t* wvb  = carve((size_t)E * E);
  bf16_t* wob  = carve((size_t)E * E);
  bf16_t* qtmp = carve((size_t)M * E);        // pre-rope projections
  bf16_t* ktmp = carve((size_t)M * E);
  bf16_t* vtmp = carve((size_t)M * E);
  bf16_t* qhB  = carve((size_t)B * H * S * HEAD_DIM);
  bf16_t* khB  = carve((size_t)B * H * S * HEAD_DIM);
  bf16_t* vTB  = carve((size_t)B * H * HEAD_DIM * S);
  bf16_t* ctx  = carve((size_t)M * E);

  // ---- stage 0: fp32 -> bf16 ---------------------------------------------
  {
    const int nx = M * E, nw = E * E;
    k_f32_to_bf16<<<(nx + 255) / 256, 256, 0, stream>>>(query, xbf, nx);
    k_f32_to_bf16<<<(nw + 255) / 256, 256, 0, stream>>>(Wq, wqb, nw);
    k_f32_to_bf16<<<(nw + 255) / 256, 256, 0, stream>>>(Wk, wkb, nw);
    k_f32_to_bf16<<<(nw + 255) / 256, 256, 0, stream>>>(Wv, wvb, nw);
    k_f32_to_bf16<<<(nw + 255) / 256, 256, 0, stream>>>(Wo, wob, nw);
  }

  // ---- stage 1: QKV projections (WMMA GEMMs) ------------------------------
  {
    dim3 grid(M / 128, E / 128);
    k_gemm_bf16<false><<<grid, 256, 0, stream>>>(xbf, wqb, bq, qtmp, M, E, E, scaling);
    k_gemm_bf16<false><<<grid, 256, 0, stream>>>(xbf, wkb, bk, ktmp, M, E, E, 1.0f);
    k_gemm_bf16<false><<<grid, 256, 0, stream>>>(xbf, wvb, bv, vtmp, M, E, E, 1.0f);
  }

  // ---- stage 2: RoPE + head scatter ---------------------------------------
  {
    const int total = S * B * H * HEAD_DIM;
    k_rope_scatter<<<(total + 255) / 256, 256, 0, stream>>>(qtmp, ktmp, vtmp,
                                                            qhB, khB, vTB);
  }

  // ---- stage 3: flash attention -------------------------------------------
  {
    dim3 grid(B * H, S / 128);
    k_flash_attn<<<grid, 256, 0, stream>>>(qhB, khB, vTB, ctx);
  }

  // ---- stage 4: output projection (fp32 out) ------------------------------
  {
    dim3 grid(M / 128, E / 128);
    k_gemm_bf16<true><<<grid, 256, 0, stream>>>(ctx, wob, bo, d_out, M, E, E, 1.0f);
  }
}
